// _ASPNet_7189775253834
// MI455X (gfx1250) — matrix operand order repaired and checked by reading, not compiled, not run on hardware
//
#include <hip/hip_runtime.h>
#include <hip/hip_bf16.h>

// ------------------------------------------------------------------
// ASPNet block for MI455X (gfx1250, wave32, WMMA).
//  GEMMs run on v_wmma_f32_16x16x32_f16 (f32 accum).
//  LDS staging uses CDNA5 GLOBAL_LOAD_ASYNC_TO_LDS_B128 (ASYNCcnt) with
//  double-buffered LDS; softmax(cov) @ g_x is algebraically collapsed to
//  softmax(cov) @ mean(g_x) since y only feeds a spatial mean.
// ------------------------------------------------------------------

typedef _Float16 f16;
typedef __attribute__((ext_vector_type(16))) _Float16 v16h;
typedef __attribute__((ext_vector_type(8)))  _Float16 v8h;
typedef __attribute__((ext_vector_type(8)))  float    v8f;
typedef __attribute__((ext_vector_type(4)))  float    v4f;

#define B_   8
#define C_   2048
#define R_   512
#define HW_  2304
#define RH_  256
#define EPS_ 1e-5f
#define SLOPE_ 0.01f

__device__ __forceinline__ float lrelu(float v) { return v >= 0.f ? v : SLOPE_ * v; }

union V16U { v16h v; v8h h[2]; };

// LDS byte offset of a generic __shared__ pointer: aperture keeps the LDS
// offset in addr[31:0] (flat->LDS mapping truncates high bits).
#define LDSOFF(p) ((unsigned)(size_t)(p))

// CDNA5 async copy: 32B global -> LDS per lane (2x b128; INST_OFFSET is added
// to BOTH the LDS and global addresses, so one address pair drives both).
__device__ __forceinline__ void async_cp32(unsigned lds_off, const f16* g) {
  asm volatile(
      "global_load_async_to_lds_b128 %0, %1, off\n\t"
      "global_load_async_to_lds_b128 %0, %1, off offset:16"
      :: "v"(lds_off), "v"(g)
      : "memory");
}
__device__ __forceinline__ void wait_async0() {
#if __has_builtin(__builtin_amdgcn_s_wait_asynccnt)
  __builtin_amdgcn_s_wait_asynccnt(0);
#else
  asm volatile("s_wait_asynccnt 0x0" ::: "memory");
#endif
}

// --- block reductions (256 threads) --------------------------------
__device__ __forceinline__ float blk_sum(float v, float* sm) {
  int t = threadIdx.x;
  sm[t] = v; __syncthreads();
  #pragma unroll
  for (int s = 128; s > 0; s >>= 1) { if (t < s) sm[t] += sm[t + s]; __syncthreads(); }
  float r = sm[0]; __syncthreads();
  return r;
}
__device__ __forceinline__ float blk_max(float v, float* sm) {
  int t = threadIdx.x;
  sm[t] = v; __syncthreads();
  #pragma unroll
  for (int s = 128; s > 0; s >>= 1) { if (t < s) sm[t] = fmaxf(sm[t], sm[t + s]); __syncthreads(); }
  float r = sm[0]; __syncthreads();
  return r;
}

// ------------------------------------------------------------------
// K0: x [B][C][HW] f32  ->  xT [B][HW][C] f16   (LDS tiled transpose)
// ------------------------------------------------------------------
__global__ __launch_bounds__(256) void k_xpose(const float* __restrict__ x,
                                               f16* __restrict__ xT) {
  __shared__ f16 tl[64][72];                 // [m][c], padded rows (144B = 9*16B)
  const int b  = blockIdx.z;
  const int c0 = blockIdx.y * 64;
  const int m0 = blockIdx.x * 64;
  const int tid = threadIdx.x;
  const int cl = tid >> 4;                   // 0..15
  const int ml = (tid & 15) * 4;             // 0..60
  #pragma unroll
  for (int i = 0; i < 4; ++i) {
    int c = cl + i * 16;
    v4f v = *(const v4f*)(x + ((size_t)(b * C_ + c0 + c)) * HW_ + m0 + ml);
    tl[ml + 0][c] = (f16)v.x; tl[ml + 1][c] = (f16)v.y;
    tl[ml + 2][c] = (f16)v.z; tl[ml + 3][c] = (f16)v.w;
  }
  __syncthreads();
  const int mr = tid >> 2;                   // 0..63
  const int cc = (tid & 3) * 16;             // 0,16,32,48
  f16* dst = xT + ((size_t)(b * HW_ + m0 + mr)) * C_ + c0 + cc;
  *(v8h*)dst       = *(const v8h*)&tl[mr][cc];
  *(v8h*)(dst + 8) = *(const v8h*)&tl[mr][cc + 8];
}

// K0b: weights f32 -> f16
__global__ void k_wconv(const float* __restrict__ w1, const float* __restrict__ w2,
                        f16* __restrict__ w1h, f16* __restrict__ w2h, int n) {
  int i = blockIdx.x * blockDim.x + threadIdx.x;
  int stride = gridDim.x * blockDim.x;
  for (; i < n; i += stride) { w1h[i] = (f16)w1[i]; w2h[i] = (f16)w2[i]; }
}

// ------------------------------------------------------------------
// K1: dual 1x1-conv GEMM.  o{1,2}[b][r][m] = sum_c w{1,2}[r][c] * x[b][c][m]
// A = w (16x32 f16 frags), B = xT rows (32x16 frags). 128x128 tile, K-step 32.
// Staging: async global->LDS, ping-pong buffers.
// ------------------------------------------------------------------
__global__ __launch_bounds__(256, 1) void k_gemm_dual(const f16* __restrict__ xT,
    const f16* __restrict__ w1h, const f16* __restrict__ w2h,
    f16* __restrict__ o1, f16* __restrict__ o2) {
  __shared__ f16 sA1[2][128 * 40];
  __shared__ f16 sA2[2][128 * 40];
  __shared__ f16 sB [2][128 * 40];
  const int b  = blockIdx.z;
  const int m0 = blockIdx.x * 128;
  const int r0 = blockIdx.y * 128;
  const int tid  = threadIdx.x;
  const int lane = tid & 31;
  const int wave = tid >> 5;
  const int wm = (wave & 1) * 64;            // wave row offset
  const int wn = (wave >> 1) * 32;           // wave col offset
  const int srow = tid >> 1;                 // staging row 0..127
  const int sch  = (tid & 1) * 16;           // 0 / 16 halves
  const int slot = srow * 40 + sch;

  const f16* gA1 = w1h + (size_t)(r0 + srow) * C_ + sch;
  const f16* gA2 = w2h + (size_t)(r0 + srow) * C_ + sch;
  const f16* gB  = xT + ((size_t)b * HW_ + m0 + srow) * C_ + sch;

  const unsigned lA1[2] = { LDSOFF(&sA1[0][slot]), LDSOFF(&sA1[1][slot]) };
  const unsigned lA2[2] = { LDSOFF(&sA2[0][slot]), LDSOFF(&sA2[1][slot]) };
  const unsigned lB [2] = { LDSOFF(&sB [0][slot]), LDSOFF(&sB [1][slot]) };

  const v8f vz = {0.f, 0.f, 0.f, 0.f, 0.f, 0.f, 0.f, 0.f};
  v8f acc[2][4][2];
  #pragma unroll
  for (int br = 0; br < 2; ++br)
    #pragma unroll
    for (int mi = 0; mi < 4; ++mi)
      #pragma unroll
      for (int n = 0; n < 2; ++n) acc[br][mi][n] = vz;

  auto issue = [&](int kt, int buf) {
    const size_t off = (size_t)kt * 32;
    async_cp32(lA1[buf], gA1 + off);
    async_cp32(lA2[buf], gA2 + off);
    async_cp32(lB [buf], gB  + off);
    if (kt + 1 < 64) {                       // -> global_prefetch_b8 (near)
      __builtin_prefetch(gB + off + 32, 0, 3);
      __builtin_prefetch(gA1 + off + 32, 0, 3);
      __builtin_prefetch(gA2 + off + 32, 0, 3);
    }
  };

  issue(0, 0);

  const int lrow = lane & 15;
  const int khB  = (lane >> 4) * 16;         // B frag: 16 contiguous halves
  const int khA  = (lane >> 4) * 16;         // A frag: 16 contiguous halves

  for (int kt = 0; kt < 64; ++kt) {
    const int cur = kt & 1;
    wait_async0();                           // this wave's slab kt landed
    __syncthreads();                         // all waves landed; prev reads done
    if (kt + 1 < 64) issue(kt + 1, cur ^ 1); // DMA next slab under compute

    const f16* tB  = sB [cur];
    const f16* tA1 = sA1[cur];
    const f16* tA2 = sA2[cur];

    V16U bfrag[2];
    #pragma unroll
    for (int n = 0; n < 2; ++n) {
      const f16* p = &tB[(wn + n * 16 + lrow) * 40 + khB];
      bfrag[n].h[0] = *(const v8h*)p; bfrag[n].h[1] = *(const v8h*)(p + 8);
    }
    #pragma unroll
    for (int mi = 0; mi < 4; ++mi) {
      const f16* pa = &tA1[(wm + mi * 16 + lrow) * 40 + khA];
      V16U ua; ua.h[0] = *(const v8h*)pa; ua.h[1] = *(const v8h*)(pa + 8);
      #pragma unroll
      for (int n = 0; n < 2; ++n)
        acc[0][mi][n] = __builtin_amdgcn_wmma_f32_16x16x32_f16(
            false, ua.v, false, bfrag[n].v, (short)0, acc[0][mi][n], false, false);
    }
    #pragma unroll
    for (int mi = 0; mi < 4; ++mi) {
      const f16* pa = &tA2[(wm + mi * 16 + lrow) * 40 + khA];
      V16U ua; ua.h[0] = *(const v8h*)pa; ua.h[1] = *(const v8h*)(pa + 8);
      #pragma unroll
      for (int n = 0; n < 2; ++n)
        acc[1][mi][n] = __builtin_amdgcn_wmma_f32_16x16x32_f16(
            false, ua.v, false, bfrag[n].v, (short)0, acc[1][mi][n], false, false);
    }
  }

  // epilogue: D layout -> lane holds col (lane&15), vgpr v holds row v (+8 hi lanes)
  #pragma unroll
  for (int br = 0; br < 2; ++br) {
    f16* o = br ? o2 : o1;
    #pragma unroll
    for (int mi = 0; mi < 4; ++mi)
      #pragma unroll
      for (int n = 0; n < 2; ++n) {
        v8f c = acc[br][mi][n];
        int col   = m0 + wn + n * 16 + lrow;
        int rbase = r0 + wm + mi * 16 + (lane >> 4) * 8;
        f16* dst = o + ((size_t)b * R_ + rbase) * HW_ + col;
        #pragma unroll
        for (int v = 0; v < 8; ++v) dst[(size_t)v * HW_] = (f16)c[v];
      }
  }
}

// ------------------------------------------------------------------
// K2: per-channel BN1/BN2 stats over (B,HW) + per-row means (gbar, mu2)
// ------------------------------------------------------------------
__global__ __launch_bounds__(256) void k_stats(const f16* __restrict__ o1,
    const f16* __restrict__ o2, float* gbar, float* mu2,
    float* bn1m, float* bn1r, float* bn2m, float* bn2r,
    float* bn3sum, float* bn3sq) {
  __shared__ float sm[256];
  const int r = blockIdx.x, t = threadIdx.x;
  float t1 = 0, tq1 = 0, t2 = 0, tq2 = 0;
  for (int b = 0; b < B_; ++b) {
    const f16* p1 = o1 + ((size_t)b * R_ + r) * HW_;
    const f16* p2 = o2 + ((size_t)b * R_ + r) * HW_;
    float s1 = 0, q1 = 0, s2 = 0, q2 = 0;
    for (int m = t; m < HW_; m += 256) {
      float v1 = (float)p1[m]; s1 += v1; q1 += v1 * v1;
      float v2 = (float)p2[m]; s2 += v2; q2 += v2 * v2;
    }
    s1 = blk_sum(s1, sm); q1 = blk_sum(q1, sm);
    s2 = blk_sum(s2, sm); q2 = blk_sum(q2, sm);
    if (t == 0) { gbar[b * R_ + r] = s1 / HW_; mu2[b * R_ + r] = s2 / HW_; }
    t1 += s1; tq1 += q1; t2 += s2; tq2 += q2;
  }
  if (t == 0) {
    const float n = (float)(B_ * HW_);
    float m1 = t1 / n, v1 = tq1 / n - m1 * m1;
    float m2 = t2 / n, v2 = tq2 / n - m2 * m2;
    bn1m[r] = m1; bn1r[r] = rsqrtf(v1 + EPS_);
    bn2m[r] = m2; bn2r[r] = rsqrtf(v2 + EPS_);
    bn3sum[r] = 0.f; bn3sq[r] = 0.f;   // reset accumulators every call
  }
}

// ------------------------------------------------------------------
// K3: xc = o2 - rowmean(o2)  (f16), and BN3 stats of p = l(bn1)*l(bn2)
// ------------------------------------------------------------------
__global__ __launch_bounds__(256) void k_xc(const f16* __restrict__ o1,
    const f16* __restrict__ o2,
    const float* g1, const float* b1, const float* g2, const float* b2,
    const float* bn1m, const float* bn1r, const float* bn2m, const float* bn2r,
    const float* mu2, f16* __restrict__ xc, float* bn3sum, float* bn3sq) {
  __shared__ float sm[256];
  const int r = blockIdx.x, b = blockIdx.y, t = threadIdx.x;
  const size_t base = ((size_t)b * R_ + r) * HW_;
  const float a1 = bn1r[r] * g1[r], c1 = b1[r] - bn1m[r] * a1;
  const float a2 = bn2r[r] * g2[r], c2 = b2[r] - bn2m[r] * a2;
  const float mu = mu2[b * R_ + r];
  float sp = 0, sq = 0;
  for (int m = t; m < HW_; m += 256) {
    float v1 = (float)o1[base + m], v2 = (float)o2[base + m];
    float p = lrelu(v1 * a1 + c1) * lrelu(v2 * a2 + c2);
    sp += p; sq += p * p;
    xc[base + m] = (f16)(v2 - mu);
  }
  sp = blk_sum(sp, sm); sq = blk_sum(sq, sm);
  if (t == 0) { atomicAdd(bn3sum + r, sp); atomicAdd(bn3sq + r, sq); }
}

__global__ void k_bn3fin(const float* bn3sum, const float* bn3sq,
                         float* bn3m, float* bn3r) {
  int r = blockIdx.x * blockDim.x + threadIdx.x;
  if (r >= R_) return;
  const float n = (float)(B_ * HW_);
  float m = bn3sum[r] / n, v = bn3sq[r] / n - m * m;
  bn3m[r] = m; bn3r[r] = rsqrtf(v + EPS_);
}

// ------------------------------------------------------------------
// K5: cov[b] = xc[b] @ xc[b]^T / HW   (WMMA f16, no transpose needed;
// async LDS staging, ping-pong)
// ------------------------------------------------------------------
__global__ __launch_bounds__(256, 1) void k_cov(const f16* __restrict__ xc,
                                                float* __restrict__ cov) {
  __shared__ f16 sA [2][128 * 40];
  __shared__ f16 sBt[2][128 * 40];
  const int b  = blockIdx.z;
  const int j0 = blockIdx.x * 128;
  const int i0 = blockIdx.y * 128;
  const int tid  = threadIdx.x;
  const int lane = tid & 31;
  const int wave = tid >> 5;
  const int wm = (wave & 1) * 64;
  const int wn = (wave >> 1) * 32;
  const int srow = tid >> 1;
  const int sch  = (tid & 1) * 16;
  const int slot = srow * 40 + sch;

  const f16* gA = xc + ((size_t)b * R_ + i0 + srow) * HW_ + sch;
  const f16* gB = xc + ((size_t)b * R_ + j0 + srow) * HW_ + sch;

  const unsigned lA[2] = { LDSOFF(&sA [0][slot]), LDSOFF(&sA [1][slot]) };
  const unsigned lB[2] = { LDSOFF(&sBt[0][slot]), LDSOFF(&sBt[1][slot]) };

  const v8f vz = {0.f, 0.f, 0.f, 0.f, 0.f, 0.f, 0.f, 0.f};
  v8f acc[4][2];
  #pragma unroll
  for (int mi = 0; mi < 4; ++mi)
    #pragma unroll
    for (int n = 0; n < 2; ++n) acc[mi][n] = vz;

  const int NK = HW_ / 32;                   // 72
  auto issue = [&](int kt, int buf) {
    const size_t off = (size_t)kt * 32;
    async_cp32(lA[buf], gA + off);
    async_cp32(lB[buf], gB + off);
    if (kt + 1 < NK) {
      __builtin_prefetch(gA + off + 32, 0, 3);
      __builtin_prefetch(gB + off + 32, 0, 3);
    }
  };

  issue(0, 0);

  const int lrow = lane & 15;
  const int khB  = (lane >> 4) * 16;
  const int khA  = (lane >> 4) * 16;

  for (int kt = 0; kt < NK; ++kt) {
    const int cur = kt & 1;
    wait_async0();
    __syncthreads();
    if (kt + 1 < NK) issue(kt + 1, cur ^ 1);

    const f16* tA = sA [cur];
    const f16* tB = sBt[cur];

    V16U bfrag[2];
    #pragma unroll
    for (int n = 0; n < 2; ++n) {
      const f16* p = &tB[(wn + n * 16 + lrow) * 40 + khB];
      bfrag[n].h[0] = *(const v8h*)p; bfrag[n].h[1] = *(const v8h*)(p + 8);
    }
    #pragma unroll
    for (int mi = 0; mi < 4; ++mi) {
      const f16* pa = &tA[(wm + mi * 16 + lrow) * 40 + khA];
      V16U ua; ua.h[0] = *(const v8h*)pa; ua.h[1] = *(const v8h*)(pa + 8);
      #pragma unroll
      for (int n = 0; n < 2; ++n)
        acc[mi][n] = __builtin_amdgcn_wmma_f32_16x16x32_f16(
            false, ua.v, false, bfrag[n].v, (short)0, acc[mi][n], false, false);
    }
  }

  const float inv = 1.0f / (float)HW_;
  #pragma unroll
  for (int mi = 0; mi < 4; ++mi)
    #pragma unroll
    for (int n = 0; n < 2; ++n) {
      v8f c = acc[mi][n];
      int col   = j0 + wn + n * 16 + lrow;
      int rbase = i0 + wm + mi * 16 + (lane >> 4) * 8;
      float* dst = cov + ((size_t)b * R_ + rbase) * R_ + col;
      #pragma unroll
      for (int v = 0; v < 8; ++v) dst[(size_t)v * R_] = c[v] * inv;
    }
}

// ------------------------------------------------------------------
// K6: per (b,i): outbar (spatial mean of fused BN3 path) + softmax row of
// cov dotted with gbar  ->  s[b,i]
// ------------------------------------------------------------------
__global__ __launch_bounds__(256) void k_smax(const float* __restrict__ cov,
    const float* __restrict__ gbar,
    const f16* __restrict__ o1, const f16* __restrict__ o2,
    const float* g1, const float* b1, const float* g2, const float* b2,
    const float* g3, const float* b3,
    const float* bn1m, const float* bn1r, const float* bn2m, const float* bn2r,
    const float* bn3m, const float* bn3r, float* __restrict__ s) {
  __shared__ float sm[256];
  const int i = blockIdx.x, b = blockIdx.y, t = threadIdx.x;
  const float a1 = bn1r[i] * g1[i], c1 = b1[i] - bn1m[i] * a1;
  const float a2 = bn2r[i] * g2[i], c2 = b2[i] - bn2m[i] * a2;
  const float a3 = bn3r[i] * g3[i], c3 = b3[i] - bn3m[i] * a3;
  const size_t base = ((size_t)b * R_ + i) * HW_;
  float so = 0.f;
  for (int m = t; m < HW_; m += 256) {
    float p = lrelu((float)o1[base + m] * a1 + c1) *
              lrelu((float)o2[base + m] * a2 + c2);
    so += lrelu(p * a3 + c3);
  }
  so = blk_sum(so, sm) / (float)HW_;

  const float* row = cov + ((size_t)b * R_ + i) * R_;
  float v0 = row[t], v1 = row[t + 256];
  float mx = blk_max(fmaxf(v0, v1), sm);
  float e0 = __expf(v0 - mx), e1 = __expf(v1 - mx);
  float se = blk_sum(e0 + e1, sm);
  const float* gb = gbar + b * R_;
  float dd = blk_sum(e0 * gb[t] + e1 * gb[t + 256], sm);
  if (t == 0) s[b * R_ + i] = dd / se + so;
}

// K7: SE gate:  h = relu(fc1 s + b1);  gate = sigmoid(fc2 h + b2)
__global__ __launch_bounds__(256) void k_se(const float* __restrict__ s,
    const float* __restrict__ fw1, const float* __restrict__ fb1,
    const float* __restrict__ fw2, const float* __restrict__ fb2,
    float* __restrict__ gate) {
  __shared__ float ss[R_];
  __shared__ float hh[RH_];
  const int b = blockIdx.x, t = threadIdx.x;
  ss[t] = s[b * R_ + t]; ss[t + 256] = s[b * R_ + t + 256];
  __syncthreads();
  float a = fb1[t];
  const float* wr = fw1 + (size_t)t * R_;
  for (int j = 0; j < R_; ++j) a += wr[j] * ss[j];
  hh[t] = fmaxf(a, 0.f);
  __syncthreads();
  for (int k = 0; k < 8; ++k) {
    int c = t + k * 256;
    const float* w2r = fw2 + (size_t)c * RH_;
    float g = fb2[c];
    for (int j = 0; j < RH_; ++j) g += w2r[j] * hh[j];
    gate[b * C_ + c] = 1.f / (1.f + __expf(-g));
  }
}

// K8: out = x * gate  (float4, coalesced)
__global__ void k_scale(const float* __restrict__ x, const float* __restrict__ gate,
                        float* __restrict__ out) {
  const int bc = blockIdx.x;
  const float g = gate[bc];
  const v4f* xi = (const v4f*)(x + (size_t)bc * HW_);
  v4f* oo = (v4f*)(out + (size_t)bc * HW_);
  for (int i = threadIdx.x; i < HW_ / 4; i += 256) {
    v4f v = xi[i];
    v.x *= g; v.y *= g; v.z *= g; v.w *= g;
    oo[i] = v;
  }
}

// ------------------------------------------------------------------
extern "C" void kernel_launch(void* const* d_in, const int* in_sizes, int n_in,
                              void* d_out, int out_size, void* d_ws, size_t ws_size,
                              hipStream_t stream) {
  const float* x   = (const float*)d_in[0];
  const float* w1  = (const float*)d_in[1];
  const float* w2  = (const float*)d_in[2];
  const float* g1  = (const float*)d_in[3];
  const float* b1  = (const float*)d_in[4];
  const float* g2  = (const float*)d_in[5];
  const float* b2  = (const float*)d_in[6];
  const float* g3  = (const float*)d_in[7];
  const float* b3  = (const float*)d_in[8];
  const float* fw1 = (const float*)d_in[9];
  const float* fb1 = (const float*)d_in[10];
  const float* fw2 = (const float*)d_in[11];
  const float* fb2 = (const float*)d_in[12];
  float* out = (float*)d_out;

  char* ws = (char*)d_ws;
  size_t off = 0;
  auto alloc = [&](size_t bytes) -> char* {
    char* p = ws + off;
    off += (bytes + 255) & ~(size_t)255;
    return p;
  };
  f16*   xT    = (f16*)alloc((size_t)B_ * HW_ * C_ * sizeof(f16));
  f16*   w1h   = (f16*)alloc((size_t)R_ * C_ * sizeof(f16));
  f16*   w2h   = (f16*)alloc((size_t)R_ * C_ * sizeof(f16));
  f16*   o1    = (f16*)alloc((size_t)B_ * R_ * HW_ * sizeof(f16));
  f16*   o2    = (f16*)alloc((size_t)B_ * R_ * HW_ * sizeof(f16));
  f16*   xc    = (f16*)alloc((size_t)B_ * R_ * HW_ * sizeof(f16));
  float* cov   = (float*)alloc((size_t)B_ * R_ * R_ * sizeof(float));
  float* gbar  = (float*)alloc((size_t)B_ * R_ * sizeof(float));
  float* mu2   = (float*)alloc((size_t)B_ * R_ * sizeof(float));
  float* sarr  = (float*)alloc((size_t)B_ * R_ * sizeof(float));
  float* bn1m  = (float*)alloc(R_ * sizeof(float));
  float* bn1r  = (float*)alloc(R_ * sizeof(float));
  float* bn2m  = (float*)alloc(R_ * sizeof(float));
  float* bn2r  = (float*)alloc(R_ * sizeof(float));
  float* bn3s  = (float*)alloc(R_ * sizeof(float));
  float* bn3q  = (float*)alloc(R_ * sizeof(float));
  float* bn3m  = (float*)alloc(R_ * sizeof(float));
  float* bn3r  = (float*)alloc(R_ * sizeof(float));
  float* gate  = (float*)alloc((size_t)B_ * C_ * sizeof(float));

  k_xpose   <<<dim3(HW_ / 64, C_ / 64, B_), 256, 0, stream>>>(x, xT);
  k_wconv   <<<dim3(512), 256, 0, stream>>>(w1, w2, w1h, w2h, R_ * C_);
  k_gemm_dual<<<dim3(HW_ / 128, R_ / 128, B_), 256, 0, stream>>>(xT, w1h, w2h, o1, o2);
  k_stats   <<<dim3(R_), 256, 0, stream>>>(o1, o2, gbar, mu2, bn1m, bn1r, bn2m, bn2r, bn3s, bn3q);
  k_xc      <<<dim3(R_, B_), 256, 0, stream>>>(o1, o2, g1, b1, g2, b2,
                                               bn1m, bn1r, bn2m, bn2r, mu2, xc, bn3s, bn3q);
  k_bn3fin  <<<dim3(2), 256, 0, stream>>>(bn3s, bn3q, bn3m, bn3r);
  k_cov     <<<dim3(R_ / 128, R_ / 128, B_), 256, 0, stream>>>(xc, cov);
  k_smax    <<<dim3(R_, B_), 256, 0, stream>>>(cov, gbar, o1, o2, g1, b1, g2, b2, g3, b3,
                                               bn1m, bn1r, bn2m, bn2r, bn3m, bn3r, sarr);
  k_se      <<<dim3(B_), 256, 0, stream>>>(sarr, fw1, fb1, fw2, fb2, gate);
  k_scale   <<<dim3(B_ * C_), 256, 0, stream>>>(x, gate, out);
}
